// PatchMatch_9938554323611
// MI455X (gfx1250) — matrix operand order repaired and checked by reading, not compiled, not run on hardware
//
#include <hip/hip_runtime.h>
#include <hip/hip_bf16.h>

// ---------------------------------------------------------------------------
// PatchMatch exact 1-NN:  Q,P in [n=2][6400 pixels][288 dims]
//   dist = ||q||^2 + ||p||^2 - 2 q.p ;  argmin over targets per query.
// GEMM via V_WMMA_F32_16X16X32_BF16 with bf16 hi/lo split (bf16x3 trick).
// ---------------------------------------------------------------------------

typedef __attribute__((ext_vector_type(16))) __bf16 v16bf;
typedef __attribute__((ext_vector_type(8)))  float  v8f;

#define NB     2
#define CH     32
#define HW     80
#define NPIX   (HW * HW)          // 6400
#define DDIM   288                // 32 * 9
#define KBLK   9                  // 288 / 32
#define NTILES 400                // 6400 / 16
#define FRAG_USHORT 512           // 32 lanes * 16 bf16 per fragment

union Frag { uint4 u[2]; v16bf v; };

__device__ __forceinline__ unsigned short f2bf_rne(float f) {
    unsigned u = __builtin_bit_cast(unsigned, f);
    unsigned r = u + 0x7FFFu + ((u >> 16) & 1u);
    return (unsigned short)(r >> 16);
}
__device__ __forceinline__ float bf2f(unsigned short h) {
    return __builtin_bit_cast(float, (unsigned)h << 16);
}

// ---------------------------------------------------------------------------
// Phase 1a: build A (query) and B (target) fragments, pre-swizzled to the
// ISA 7.12.2 VGPR layouts, split into bf16 hi + lo residual.
//   A 16x32 bf16: lane l (M=l%16), slot e: K = (e/8)*16 + (l>=16?8:0) + e%8
//   B 32x16 bf16: lane l (N=l%16), slot e: K = (e/8)*16 + (l>=16?8:0) + e%8
// ---------------------------------------------------------------------------
__global__ __launch_bounds__(256) void build_frags(
    const float* __restrict__ src, const float* __restrict__ tgt,
    unsigned short* __restrict__ Ahi, unsigned short* __restrict__ Alo,
    unsigned short* __restrict__ Bhi, unsigned short* __restrict__ Blo)
{
    unsigned tid = blockIdx.x * 256u + threadIdx.x;
    const unsigned total = 2u * NB * NTILES * KBLK * FRAG_USHORT;  // 7,372,800
    if (tid >= total) return;

    unsigned e  = tid & 15u;
    unsigned l  = (tid >> 4) & 31u;
    unsigned r  = tid >> 9;
    unsigned kb = r % KBLK;   r /= KBLK;
    unsigned tl = r % NTILES; r /= NTILES;
    unsigned nidx  = r & 1u;
    unsigned which = r >> 1;  // 0 = A/source, 1 = B/target

    unsigned kw = which ? ((e >> 3) * 16u + ((l >= 16u) ? 8u : 0u) + (e & 7u))
                        : ((e >> 3) * 16u + ((l >= 16u) ? 8u : 0u) + (e & 7u));
    unsigned K   = kb * 32u + kw;          // 0..287
    unsigned pix = tl * 16u + (l & 15u);   // 0..6399
    int y = (int)(pix / HW), x = (int)(pix % HW);
    unsigned c = K / 9u, p9 = K % 9u;
    int yy = y + (int)(p9 / 3u) - 1;
    int xx = x + (int)(p9 % 3u) - 1;

    const float* map = which ? tgt : src;
    float val = 0.f;
    if (yy >= 0 && yy < HW && xx >= 0 && xx < HW)
        val = map[((nidx * CH + c) * HW + (unsigned)yy) * HW + (unsigned)xx];

    unsigned short h  = f2bf_rne(val);
    unsigned short lo = f2bf_rne(val - bf2f(h));
    size_t off = ((size_t)(nidx * NTILES + tl) * KBLK + kb) * FRAG_USHORT + l * 16u + e;
    if (which) { Bhi[off] = h; Blo[off] = lo; }
    else       { Ahi[off] = h; Alo[off] = lo; }
}

// ---------------------------------------------------------------------------
// Phase 1b: exact fp32 squared norms of every descriptor.
// ---------------------------------------------------------------------------
__global__ __launch_bounds__(256) void build_norms(
    const float* __restrict__ src, const float* __restrict__ tgt,
    float* __restrict__ q2, float* __restrict__ p2)
{
    unsigned tid = blockIdx.x * 256u + threadIdx.x;
    if (tid >= 2u * NB * NPIX) return;
    unsigned s     = tid % NPIX;
    unsigned nidx  = (tid / NPIX) & 1u;
    unsigned which = tid / (NB * NPIX);
    const float* map = which ? tgt : src;
    int y = (int)(s / HW), x = (int)(s % HW);
    float acc = 0.f;
    for (int c = 0; c < CH; ++c)
        for (int dy = -1; dy <= 1; ++dy)
            for (int dx = -1; dx <= 1; ++dx) {
                int yy = y + dy, xx = x + dx;
                if (yy >= 0 && yy < HW && xx >= 0 && xx < HW) {
                    float v = map[((nidx * CH + c) * HW + yy) * HW + xx];
                    acc = fmaf(v, v, acc);
                }
            }
    (which ? p2 : q2)[nidx * NPIX + s] = acc;
}

// ---------------------------------------------------------------------------
// Phase 2: fused GEMM + argmin.
// Block = 256 threads = 8 waves; 32-row A strip staged in LDS (hi+lo, 36 KB).
// Wave (wm = w&1, wn = w>>2bits): M tile = strip*2+wm, two N tiles per wave.
// bf16x3: acc += Ahi*Bhi + Ahi*Blo + Alo*Bhi  (3 WMMAs per K-block per acc).
// ---------------------------------------------------------------------------
__global__ __launch_bounds__(256) void gemm_argmin(
    const unsigned short* __restrict__ Ahi, const unsigned short* __restrict__ Alo,
    const unsigned short* __restrict__ Bhi, const unsigned short* __restrict__ Blo,
    const float* __restrict__ q2, const float* __restrict__ p2,
    float* __restrict__ out)
{
    __shared__ __align__(16) unsigned short sA[2 * 2 * KBLK * FRAG_USHORT]; // hi|lo blocks

    int blk   = blockIdx.x;          // 0..399
    int nidx  = blk / 200;
    int strip = blk % 200;           // 32-row strip
    int tIdx  = threadIdx.x;
    int w     = tIdx >> 5;
    int lane  = tIdx & 31;

    // Stage A strip (2 mtiles x 9 kblocks x 512 ushort, hi and lo) into LDS.
    {
        size_t gbase = ((size_t)(nidx * NTILES + strip * 2) * KBLK) * FRAG_USHORT;
        const uint4* gh = (const uint4*)(Ahi + gbase);
        const uint4* gl = (const uint4*)(Alo + gbase);
        uint4* sh = (uint4*)sA;              // 1152 uint4 of hi
        uint4* sl = ((uint4*)sA) + 1152;     // 1152 uint4 of lo
        for (int i = tIdx; i < 1152; i += 256) { sh[i] = gh[i]; sl[i] = gl[i]; }
    }
    __syncthreads();

    int wm = w & 1;        // M tile within strip
    int wn = w >> 1;       // 0..3 : N group
    int mtile = strip * 2 + wm;

    const unsigned short* sAhi = sA + (size_t)wm * KBLK * FRAG_USHORT;
    const unsigned short* sAlo = sA + (size_t)(2 + wm) * KBLK * FRAG_USHORT;

    float bestd[8];
    int   besti[8];
#pragma unroll
    for (int r = 0; r < 8; ++r) { bestd[r] = 3.4e38f; besti[r] = 0; }

    for (int tt = 0; tt < 50; ++tt) {     // 50 iterations x 128 columns
        int nt0 = tt * 8 + wn * 2;        // first of two 16-col N tiles
        if (tt + 1 < 50)                  // gfx1250 global_prefetch_b8
            __builtin_prefetch(Bhi + ((size_t)(nidx * NTILES + nt0 + 8) * KBLK) * FRAG_USHORT, 0, 1);

        v8f acc0 = {};
        v8f acc1 = {};
#pragma unroll
        for (int kb = 0; kb < KBLK; ++kb) {
            Frag ah, al, b0h, b0l, b1h, b1l;
            const uint4* pa = (const uint4*)(sAhi + kb * FRAG_USHORT + lane * 16);
            ah.u[0] = pa[0]; ah.u[1] = pa[1];
            const uint4* pl = (const uint4*)(sAlo + kb * FRAG_USHORT + lane * 16);
            al.u[0] = pl[0]; al.u[1] = pl[1];

            size_t b0 = ((size_t)(nidx * NTILES + nt0) * KBLK + kb) * FRAG_USHORT + lane * 16;
            size_t b1 = b0 + (size_t)KBLK * FRAG_USHORT;
            const uint4* p0h = (const uint4*)(Bhi + b0); b0h.u[0] = p0h[0]; b0h.u[1] = p0h[1];
            const uint4* p0l = (const uint4*)(Blo + b0); b0l.u[0] = p0l[0]; b0l.u[1] = p0l[1];
            const uint4* p1h = (const uint4*)(Bhi + b1); b1h.u[0] = p1h[0]; b1h.u[1] = p1h[1];
            const uint4* p1l = (const uint4*)(Blo + b1); b1l.u[0] = p1l[0]; b1l.u[1] = p1l[1];

            acc0 = __builtin_amdgcn_wmma_f32_16x16x32_bf16(false, ah.v, false, b0h.v, (short)0, acc0, false, false);
            acc0 = __builtin_amdgcn_wmma_f32_16x16x32_bf16(false, ah.v, false, b0l.v, (short)0, acc0, false, false);
            acc0 = __builtin_amdgcn_wmma_f32_16x16x32_bf16(false, al.v, false, b0h.v, (short)0, acc0, false, false);
            acc1 = __builtin_amdgcn_wmma_f32_16x16x32_bf16(false, ah.v, false, b1h.v, (short)0, acc1, false, false);
            acc1 = __builtin_amdgcn_wmma_f32_16x16x32_bf16(false, ah.v, false, b1l.v, (short)0, acc1, false, false);
            acc1 = __builtin_amdgcn_wmma_f32_16x16x32_bf16(false, al.v, false, b1h.v, (short)0, acc1, false, false);
        }

        // Epilogue: track min of (p^2 - 2*dot); q^2 (row-constant) added at end.
        int col0 = nt0 * 16 + (lane & 15);
        float p20 = p2[nidx * NPIX + col0];
        float p21 = p2[nidx * NPIX + col0 + 16];
#pragma unroll
        for (int r = 0; r < 8; ++r) {
            float d0 = fmaf(acc0[r], -2.f, p20);
            if (d0 < bestd[r]) { bestd[r] = d0; besti[r] = col0; }
            float d1 = fmaf(acc1[r], -2.f, p21);
            if (d1 < bestd[r]) { bestd[r] = d1; besti[r] = col0 + 16; }
        }
    }

    // Reduce across the 16 lanes holding each row (halves stay disjoint).
#pragma unroll
    for (int off = 1; off < 16; off <<= 1) {
#pragma unroll
        for (int r = 0; r < 8; ++r) {
            float od = __shfl_xor(bestd[r], off, 32);
            int   oi = __shfl_xor(besti[r], off, 32);
            if (od < bestd[r] || (od == bestd[r] && oi < besti[r])) {
                bestd[r] = od; besti[r] = oi;
            }
        }
    }

    // Cross-wave (N-group) reduction through LDS, then write results.
    __syncthreads();
    float* redD = (float*)sA;             // [3][32]
    int*   redI = (int*)(redD + 3 * 32);  // [3][32]
    if (wn > 0 && (lane & 15) == 0) {
#pragma unroll
        for (int r = 0; r < 8; ++r) {
            int row = wm * 16 + r + 8 * (lane >> 4);
            redD[(wn - 1) * 32 + row] = bestd[r];
            redI[(wn - 1) * 32 + row] = besti[r];
        }
    }
    __syncthreads();
    if (wn == 0 && (lane & 15) == 0) {
#pragma unroll
        for (int r = 0; r < 8; ++r) {
            int row = wm * 16 + r + 8 * (lane >> 4);
            float bd = bestd[r]; int bi = besti[r];
            for (int k = 0; k < 3; ++k) {
                float od = redD[k * 32 + row]; int oi = redI[k * 32 + row];
                if (od < bd || (od == bd && oi < bi)) { bd = od; bi = oi; }
            }
            int srow = mtile * 16 + r + 8 * (lane >> 4);
            float dmin = bd + q2[nidx * NPIX + srow];
            out[(size_t)(nidx * 2 + 0) * NPIX + srow] = (float)(bi / HW);
            out[(size_t)(nidx * 2 + 1) * NPIX + srow] = (float)(bi % HW);
            out[(size_t)NB * 2 * NPIX + nidx * NPIX + srow] = dmin;
        }
    }
}

// ---------------------------------------------------------------------------
// Fallback (workspace too small): exact brute force, one thread per query.
// ---------------------------------------------------------------------------
__global__ void naive_nn(const float* __restrict__ src, const float* __restrict__ tgt,
                         float* __restrict__ out)
{
    int gid = blockIdx.x * blockDim.x + threadIdx.x;
    if (gid >= NB * NPIX) return;
    int nidx = gid / NPIX, sp = gid % NPIX;
    int ys = sp / HW, xs = sp % HW;
    float q[DDIM];
    int d = 0;
    for (int c = 0; c < CH; ++c)
        for (int dy = -1; dy <= 1; ++dy)
            for (int dx = -1; dx <= 1; ++dx) {
                int yy = ys + dy, xx = xs + dx;
                q[d++] = (yy >= 0 && yy < HW && xx >= 0 && xx < HW)
                         ? src[((nidx * CH + c) * HW + yy) * HW + xx] : 0.f;
            }
    float best = 3.4e38f; int bi = 0;
    for (int t = 0; t < NPIX; ++t) {
        int yt = t / HW, xt = t % HW;
        float acc = 0.f; d = 0;
        for (int c = 0; c < CH; ++c)
            for (int dy = -1; dy <= 1; ++dy)
                for (int dx = -1; dx <= 1; ++dx) {
                    int yy = yt + dy, xx = xt + dx;
                    float pv = (yy >= 0 && yy < HW && xx >= 0 && xx < HW)
                               ? tgt[((nidx * CH + c) * HW + yy) * HW + xx] : 0.f;
                    float df = q[d++] - pv;
                    acc = fmaf(df, df, acc);
                }
        if (acc < best) { best = acc; bi = t; }
    }
    out[(nidx * 2 + 0) * NPIX + sp] = (float)(bi / HW);
    out[(nidx * 2 + 1) * NPIX + sp] = (float)(bi % HW);
    out[NB * 2 * NPIX + nidx * NPIX + sp] = best;
}

extern "C" void kernel_launch(void* const* d_in, const int* in_sizes, int n_in,
                              void* d_out, int out_size, void* d_ws, size_t ws_size,
                              hipStream_t stream) {
    const float* src = (const float*)d_in[0];
    const float* tgt = (const float*)d_in[1];
    float* out = (float*)d_out;

    const size_t fragElems = (size_t)NB * NTILES * KBLK * FRAG_USHORT; // 3,686,400 ushort
    const size_t needed = fragElems * 2 * 4 + (size_t)2 * NB * NPIX * 4;
    if (ws_size < needed) {
        naive_nn<<<(NB * NPIX + 255) / 256, 256, 0, stream>>>(src, tgt, out);
        return;
    }

    unsigned short* Ahi = (unsigned short*)d_ws;
    unsigned short* Alo = Ahi + fragElems;
    unsigned short* Bhi = Alo + fragElems;
    unsigned short* Blo = Bhi + fragElems;
    float* q2 = (float*)(Blo + fragElems);
    float* p2 = q2 + (size_t)NB * NPIX;

    const unsigned buildThreads = 2u * NB * NTILES * KBLK * FRAG_USHORT;
    build_frags<<<(buildThreads + 255) / 256, 256, 0, stream>>>(src, tgt, Ahi, Alo, Bhi, Blo);
    build_norms<<<(2 * NB * NPIX + 255) / 256, 256, 0, stream>>>(src, tgt, q2, p2);
    gemm_argmin<<<NB * 200, 256, 0, stream>>>(Ahi, Alo, Bhi, Blo, q2, p2, out);
}
